// TimeLSTM_7559142441745
// MI455X (gfx1250) — matrix operand order repaired and checked by reading, not compiled, not run on hardware
//
#include <hip/hip_runtime.h>

// ---------------------------------------------------------------------------
// TimeLSTM for MI455X (gfx1250), wave32 + WMMA bf16 (f32 accumulate).
//
// Kernel 1: fold weights -> bf16 workspace:
//   Wcat[g][k], g in [0,1024), k in [0,512): k<256 -> W_all_w[g][k] (h-part),
//                                            k>=256 -> U_all_w[g][k-256] (x-part)
//   WdB[n][k] = W_d_w[n][k] (bf16), ball[g] = W_all_b[g] + U_all_b[g]
//
// Kernel 2: persistent scan. 8 workgroups x 512 threads (16 waves).
//   Block b owns batch rows [16b, 16b+16). Wave w owns hidden cols [16w,16w+16).
//   Per step: outs = [h|x] @ Wcat^T  (K=512, 4 gate tiles per wave, WMMA)
//             d    = c @ Wd^T        (K=256, 1 tile per wave, WMMA)
//   x-staging for step s+1 is software-pipelined across step s's GEMM via a
//   double-buffered [h|x] LDS image, so no memory-latency segment sits on the
//   1024-step serial chain. Elementwise gate math is fully in-lane because the
//   WMMA C/D layout (VGPR r <-> row r+8*(lane>=16), col = lane&15) coincides
//   with the lane's ownership map; the f32 cell state lives in 8 registers.
// ---------------------------------------------------------------------------

typedef __bf16 bf16_t;
typedef __attribute__((ext_vector_type(16))) __bf16 v16bf;
typedef __attribute__((ext_vector_type(8)))  __bf16 v8bf;
typedef __attribute__((ext_vector_type(8)))  float  v8f;
typedef __attribute__((ext_vector_type(4)))  float  v4f;

#define S_LEN 1024
#define HID   256
#define GATES 1024   // 4*HID
#define KTOT  512    // HID (h-part) + HID (x-part)

__device__ __forceinline__ float sigmoid_f(float x) {
  return 1.0f / (1.0f + __expf(-x));
}

__global__ void cvt_weights_kernel(const float* __restrict__ Wall,
                                   const float* __restrict__ Wallb,
                                   const float* __restrict__ Uall,
                                   const float* __restrict__ Uallb,
                                   const float* __restrict__ Wd,
                                   bf16_t* __restrict__ WcatB,
                                   bf16_t* __restrict__ WdB,
                                   float* __restrict__ ball) {
  const int stride = gridDim.x * blockDim.x;
  const int tid0 = blockIdx.x * blockDim.x + threadIdx.x;
  for (int idx = tid0; idx < GATES * KTOT; idx += stride) {
    const int g = idx >> 9;           // / KTOT
    const int k = idx & (KTOT - 1);
    const float v = (k < HID) ? Wall[g * HID + k] : Uall[g * HID + (k - HID)];
    WcatB[idx] = (bf16_t)v;
  }
  for (int idx = tid0; idx < HID * HID; idx += stride)
    WdB[idx] = (bf16_t)Wd[idx];
  for (int idx = tid0; idx < GATES; idx += stride)
    ball[idx] = Wallb[idx] + Uallb[idx];
}

__device__ __forceinline__ v8bf pack_bf8(const v4f a, const v4f b) {
  v8bf xv;
  xv[0] = (bf16_t)a[0]; xv[1] = (bf16_t)a[1];
  xv[2] = (bf16_t)a[2]; xv[3] = (bf16_t)a[3];
  xv[4] = (bf16_t)b[0]; xv[5] = (bf16_t)b[1];
  xv[6] = (bf16_t)b[2]; xv[7] = (bf16_t)b[3];
  return xv;
}

__launch_bounds__(512, 1)
__global__ void timelstm_scan_kernel(const float* __restrict__ xin,   // [B,S,H] f32
                                     const float* __restrict__ ts,    // [B,S]   f32
                                     const bf16_t* __restrict__ Wcat, // [1024][512] bf16
                                     const bf16_t* __restrict__ WdB,  // [256][256]  bf16
                                     const float* __restrict__ ball,  // [1024] f32
                                     const float* __restrict__ bd,    // [256]  f32
                                     float* __restrict__ out) {       // [B,S,H] f32
  // double-buffered [h | x] A-operand image (cols 0..255 = h, 256..511 = x)
  __shared__ __align__(16) bf16_t hx[2][16][KTOT];
  __shared__ __align__(16) bf16_t c_bf[16][HID];
  __shared__ float t_l[2][16];

  const int tid  = threadIdx.x;
  const int lane = tid & 31;
  const int wave = tid >> 5;          // 0..15 : owns hidden-column block
  const int b0   = blockIdx.x * 16;   // batch-row tile

  const int mlo = lane & 15;          // A row / B,C,D column selector
  const int khi = lane >> 4;          // 0/1 : K-half (A,B) / M-half (C,D)

  // staging map: thread -> 8 contiguous elements of one row of x_s
  const int m_st = tid >> 5;          // row  (== wave)
  const int k_st = (tid & 31) * 8;    // col offset within the row

  // --- init: h = c = 0 (zero both hx buffers + c_bf) ---
  for (int i = tid; i < 2 * 16 * KTOT; i += 512) (&hx[0][0][0])[i] = (bf16_t)0.0f;
  for (int i = tid; i < 16 * HID; i += 512)      (&c_bf[0][0])[i]  = (bf16_t)0.0f;
  float creg[8];                       // f32 master cell state, lane-owned
#pragma unroll
  for (int r = 0; r < 8; ++r) creg[r] = 0.0f;

  // --- prologue: stage x(0) + t(0) into buffer 0 ---
  {
    const float* src = xin + (size_t)(b0 + m_st) * (S_LEN * HID) + k_st;
    const v4f a = *(const v4f*)src;
    const v4f b = *(const v4f*)(src + 4);
    *(v8bf*)&hx[0][m_st][HID + k_st] = pack_bf8(a, b);
    if (tid < 16) t_l[0][tid] = ts[(size_t)(b0 + tid) * S_LEN];
  }
  __syncthreads();

  // step-invariant bias loads for owned columns
  const int j = wave * 16 + mlo;
  const float bf_g = ball[0 * HID + j];
  const float bi_g = ball[1 * HID + j];
  const float bo_g = ball[2 * HID + j];
  const float bc_g = ball[3 * HID + j];
  const float bd_j = bd[j];

  for (int s = 0; s < S_LEN; ++s) {
    const int cur = s & 1;
    const int nxt = cur ^ 1;
    const bool have_next = (s + 1 < S_LEN);

    // ---- prefetch x(s+1) into registers; rides the WMMA shadow ----
    v4f pxa = {}, pxb = {};
    if (have_next) {
      const float* src = xin + (size_t)(b0 + m_st) * (S_LEN * HID) +
                         (size_t)(s + 1) * HID + k_st;
      pxa = *(const v4f*)src;
      pxb = *(const v4f*)(src + 4);
    }

    // ---- WMMA GEMMs (read hx[cur], c_bf) ----
    v8f accG[4] = {};   // 4 gate tiles for this wave's column block
    v8f accD   = {};    // W_d tile

    const bf16_t* Abase = &hx[cur][0][0];   // single base: no per-kt select
#pragma unroll 4
    for (int kt = 0; kt < 16; ++kt) {
      const int k0 = kt * 32;
      const bf16_t* arow = Abase + mlo * KTOT + k0 + khi * 16;
      const v8bf alo = *(const v8bf*)(arow);        // K {k0+kb .. +7}
      const v8bf ahi = *(const v8bf*)(arow + 8);    // K {k0+kb+8 .. +15}
      const v16bf a = __builtin_shufflevector(
          alo, ahi, 0, 1, 2, 3, 4, 5, 6, 7, 8, 9, 10, 11, 12, 13, 14, 15);
#pragma unroll
      for (int g = 0; g < 4; ++g) {
        const int n = g * HID + j;                  // gate column, B[k][n] = Wcat[n][k]
        const v16bf bfrag =
            *(const v16bf*)(Wcat + (size_t)n * KTOT + k0 + khi * 16);
        accG[g] = __builtin_amdgcn_wmma_f32_16x16x32_bf16(
            false, a, false, bfrag, (short)0, accG[g], false, false);
      }
    }

#pragma unroll 4
    for (int kt = 0; kt < 8; ++kt) {
      const int k0 = kt * 32;
      const bf16_t* arow = &c_bf[mlo][k0 + khi * 16];
      const v8bf alo = *(const v8bf*)(arow);
      const v8bf ahi = *(const v8bf*)(arow + 8);
      const v16bf a = __builtin_shufflevector(
          alo, ahi, 0, 1, 2, 3, 4, 5, 6, 7, 8, 9, 10, 11, 12, 13, 14, 15);
      const v16bf bfrag =
          *(const v16bf*)(WdB + (size_t)j * HID + k0 + khi * 16);
      accD = __builtin_amdgcn_wmma_f32_16x16x32_bf16(
          false, a, false, bfrag, (short)0, accD, false, false);
    }

    // ---- commit staged x(s+1) / t(s+1) into the other buffer ----
    if (have_next) {
      *(v8bf*)&hx[nxt][m_st][HID + k_st] = pack_bf8(pxa, pxb);
      if (tid < 16) t_l[nxt][tid] = ts[(size_t)(b0 + tid) * S_LEN + (s + 1)];
    }
    __syncthreads();   // GEMM reads of hx[cur]/c_bf done; x(s+1) staged

    // ---- elementwise gate math + state update (fully in-lane) ----
#pragma unroll
    for (int r = 0; r < 8; ++r) {
      const int m = r + khi * 8;
      const float c1   = tanhf(accD[r] + bd_j);          // c_s1
      const float t    = t_l[cur][m];
      const float cold = creg[r];
      const float cadj = (cold - c1) + c1 * t;
      const float fg = sigmoid_f(accG[0][r] + bf_g);
      const float ig = sigmoid_f(accG[1][r] + bi_g);
      const float og = sigmoid_f(accG[2][r] + bo_g);
      const float cg = sigmoid_f(accG[3][r] + bc_g);
      const float cn = fg * cadj + ig * cg;
      const float hn = og * tanhf(cn);
      creg[r]     = cn;
      c_bf[m][j]  = (bf16_t)cn;
      hx[nxt][m][j] = (bf16_t)hn;                        // h for next step's GEMM
      out[(size_t)(b0 + m) * (S_LEN * HID) + (size_t)s * HID + j] = hn;
    }
    __syncthreads();   // new h/c fully visible before next step's GEMMs
  }
}

extern "C" void kernel_launch(void* const* d_in, const int* in_sizes, int n_in,
                              void* d_out, int out_size, void* d_ws, size_t ws_size,
                              hipStream_t stream) {
  const float* xin   = (const float*)d_in[0];  // inputs     [128,1024,256]
  const float* ts    = (const float*)d_in[1];  // timestamps [128,1024]
  const float* Wall  = (const float*)d_in[2];  // W_all_w    [1024,256]
  const float* Wallb = (const float*)d_in[3];  // W_all_b    [1024]
  const float* Uall  = (const float*)d_in[4];  // U_all_w    [1024,256]
  const float* Uallb = (const float*)d_in[5];  // U_all_b    [1024]
  const float* Wd    = (const float*)d_in[6];  // W_d_w      [256,256]
  const float* Wdb   = (const float*)d_in[7];  // W_d_b      [256]
  float* out = (float*)d_out;

  // workspace layout: Wcat bf16 [1024*512] | Wd bf16 [256*256] | ball f32 [1024]
  bf16_t* WcatB = (bf16_t*)d_ws;
  bf16_t* WdB   = WcatB + (size_t)GATES * KTOT;
  float*  ball  = (float*)(WdB + (size_t)HID * HID);

  cvt_weights_kernel<<<1024, 256, 0, stream>>>(Wall, Wallb, Uall, Uallb, Wd,
                                               WcatB, WdB, ball);
  timelstm_scan_kernel<<<8, 512, 0, stream>>>(xin, ts, WcatB, WdB, ball, Wdb, out);
}
